// U_Net_56891136803207
// MI455X (gfx1250) — matrix operand order repaired and checked by reading, compiled, not run on hardware
//
#include <hip/hip_runtime.h>
#include <math.h>
#include <stdint.h>

typedef __attribute__((ext_vector_type(16))) _Float16 v16h;
typedef __attribute__((ext_vector_type(8)))  float    v8f;
typedef _Float16 h16;

// ---------------------------------------------------------------------------
// Elementwise kernels
// ---------------------------------------------------------------------------
__global__ void k_f32_to_f16(const float* __restrict__ in, h16* __restrict__ out, long n) {
  long i = (long)blockIdx.x * blockDim.x + threadIdx.x;
  if (i < n) out[i] = (h16)in[i];
}

__global__ void k_add_f16(const h16* __restrict__ a, const h16* __restrict__ b,
                          h16* __restrict__ o, long n) {
  long i = (long)blockIdx.x * blockDim.x + threadIdx.x;
  if (i < n) o[i] = (h16)((float)a[i] + (float)b[i]);
}

__global__ void k_final_add(const h16* __restrict__ a, const float* __restrict__ x,
                            float* __restrict__ o, long n) {
  long i = (long)blockIdx.x * blockDim.x + threadIdx.x;
  if (i < n) o[i] = (float)a[i] + x[i];
}

// ---------------------------------------------------------------------------
// Weight packing: f32 conv (OIDHW) or deconv (IODHW, flip+transpose) into
// padded f16 GEMM A-matrix [Mpad][Kpad], K = ci*K3^3 + kd*K3^2 + kh*K3 + kw.
// ---------------------------------------------------------------------------
__global__ void k_pack_w(const float* __restrict__ w, h16* __restrict__ A,
                         int O, int I, int K3, int Mpad, int Kpad, int deconv) {
  long idx = (long)blockIdx.x * blockDim.x + threadIdx.x;
  long tot = (long)Mpad * Kpad;
  if (idx >= tot) return;
  int m = (int)(idx / Kpad);
  int k = (int)(idx % Kpad);
  int KHW = K3 * K3, KV = K3 * KHW;
  int Kr = I * KV;
  float v = 0.f;
  if (m < O && k < Kr) {
    int ci = k / KV;   int r  = k - ci * KV;
    int kd = r / KHW;  int r2 = r - kd * KHW;
    int kh = r2 / K3;  int kw = r2 - kh * K3;
    if (!deconv)
      v = w[(((long)m * I + ci) * K3 + kd) * KHW + kh * K3 + kw];
    else
      v = w[(((long)ci * O + m) * K3 + (K3 - 1 - kd)) * KHW
            + (K3 - 1 - kh) * K3 + (K3 - 1 - kw)];
  }
  A[idx] = (h16)v;
}

// ---------------------------------------------------------------------------
// Implicit-GEMM conv3d via V_WMMA_F32_16X16X32_F16.
//   C[m][n] = sum_k A[m][k] * B[k][n],  m = Cout, n = Do*Ho*Wo, k = Cin*K3^3
// Block: 128 threads (4 wave32). Tile 64(M) x 64(N), K-step 32.
//  - A fragments loaded DIRECTLY from global: per ISA 7.12.2 the 16x32 f16
//    A layout is two aligned 16B chunks per lane of the packed weight row.
//  - B (im2col gather) double-buffered in LDS, stored [n][k] so each lane
//    reads aligned u32 K-pairs. One barrier per K-step.
//  - K3/PAD are template constants so the gather index math uses
//    constant-divisor strength reduction (div 27 / div 9) or vanishes (K3=1).
// ---------------------------------------------------------------------------
#define TM 64
#define TN 64
#define TK 32

union FragAB { v16h v; uint32_t u[8]; uint4 q[2]; };
union FragC  { v8f  v; float   f[8]; };

template <int K3, int PAD>
__global__ __launch_bounds__(128) void k_conv_wmma(
    const h16* __restrict__ in,   // [Cin][Din][Hin][Win] f16
    const h16* __restrict__ A,    // [Mpad][Kpad] f16 packed weights
    const float* __restrict__ bias,
    h16* __restrict__ out,        // [Cout][Do][Ho][Wo] f16
    int Din, int Hin, int Win,
    int Cout, int Do, int Ho, int Wo,
    int sH, int sW,
    int Kreal, int Kpad) {
  __shared__ __align__(16) h16 ldsB[2][TN * TK];   // [n][k], double buffered

  const int t    = threadIdx.x;
  const int wave = t >> 5;
  const int lane = t & 31;
  const long Ntot = (long)Do * Ho * Wo;
  const long n0   = (long)blockIdx.x * TN;
  const int  m0   = blockIdx.y * TM;
  const long inDHW = (long)Din * Hin * Win;
  const long inHW  = (long)Hin * Win;
  constexpr int KHW = K3 * K3;
  constexpr int KV  = K3 * KHW;

  FragC acc[4];
#pragma unroll
  for (int j = 0; j < 4; ++j)
#pragma unroll
    for (int r = 0; r < 8; ++r) acc[j].f[r] = 0.f;

  // B-fill assignment: thread handles n = t/2, k range [(t&1)*16, +16)
  const int bn = t >> 1;
  const int kb = (t & 1) * 16;
  const long ng = n0 + bn;
  const bool nok = ng < Ntot;
  int wo = 0, ho = 0, dd = 0;
  if (nok) {
    wo = (int)(ng % Wo);
    long r = ng / Wo;
    ho = (int)(r % Ho);
    dd = (int)(r / Ho);
  }

  auto fillB = [&](h16* dst, int k0) {
#pragma unroll 4
    for (int i = 0; i < 16; ++i) {
      int kg = k0 + kb + i;
      float v = 0.f;
      if (nok && kg < Kreal) {
        int ci = kg / KV;  int r  = kg - ci * KV;
        int kd = r / KHW;  int r2 = r - kd * KHW;
        int kh = r2 / K3;  int kw = r2 - kh * K3;
        int id = dd + kd - PAD;        // strideD == 1 everywhere
        int ih = ho * sH + kh - PAD;
        int iw = wo * sW + kw - PAD;
        if ((unsigned)id < (unsigned)Din && (unsigned)ih < (unsigned)Hin &&
            (unsigned)iw < (unsigned)Win)
          v = (float)in[(long)ci * inDHW + (long)id * inHW + (long)ih * Win + iw];
      }
      dst[bn * TK + kb + i] = (h16)v;
    }
  };

  // A fragment source: lane holds row m = m0 + wave*16 + (lane&15),
  // 16B chunks {g, 2+g} of the 32-half K-slab, g = (lane>=16).
  const uint4* arow =
      (const uint4*)(A + (long)(m0 + (wave << 4) + (lane & 15)) * Kpad);
  const int g = (lane >= 16) ? 1 : 0;

  fillB(ldsB[0], 0);
  __syncthreads();

  int p = 0;
  for (int k0 = 0; k0 < Kpad; k0 += TK) {
    // ---- A fragment direct from global (L2-resident packed weights) ----
    FragAB a;
    {
      const uint4* s = arow + (k0 >> 3);   // 8 halfs per uint4
      a.q[0] = s[g];
      a.q[1] = s[2 + g];
    }
    // ---- 4 WMMAs against the current B buffer ----
    {
      const uint32_t* lB = (const uint32_t*)ldsB[p];
      const int koffB = (lane >= 16) ? 8 : 0;
#pragma unroll
      for (int j = 0; j < 4; ++j) {
        FragAB b;
        int nl = (j << 4) + (lane & 15);
#pragma unroll
        for (int v = 0; v < 8; ++v) b.u[v] = lB[(nl * TK + 2 * v + (v >= 4 ? 8 : 0) + koffB) >> 1];
        acc[j].v = __builtin_amdgcn_wmma_f32_16x16x32_f16(
            false, a.v, false, b.v, (short)0, acc[j].v, false, false);
      }
    }
    // ---- prefetch-gather next B tile into the other buffer ----
    if (k0 + TK < Kpad) fillB(ldsB[p ^ 1], k0 + TK);
    __syncthreads();
    p ^= 1;
  }

  // ---- epilogue: C layout = lanes 0-15 M=r, lanes 16-31 M=8+r ----
  int mbase = m0 + (wave << 4) + ((lane >= 16) ? 8 : 0);
  int ncol  = lane & 15;
#pragma unroll
  for (int j = 0; j < 4; ++j) {
    long n = n0 + (j << 4) + ncol;
    if (n >= Ntot) continue;
#pragma unroll
    for (int r = 0; r < 8; ++r) {
      int m = mbase + r;
      if (m < Cout) {
        float v = acc[j].f[r];
        if (bias) v += bias[m];
        out[(long)m * Ntot + n] = (h16)v;
      }
    }
  }
}

// ---------------------------------------------------------------------------
// QRNN forget-gate scan over D. gates [2C][D][HW]: z then f. Out [C][D][HW].
// ---------------------------------------------------------------------------
__global__ void k_qrnn2(const h16* __restrict__ g, h16* __restrict__ out,
                        int C, int D, long HW, int reverse) {
  long i = (long)blockIdx.x * blockDim.x + threadIdx.x;
  if (i >= (long)C * HW) return;
  int c = (int)(i / HW);
  long hw = i % HW;
  long DHW = (long)D * HW;
  const h16* gz = g + (long)c * DHW + hw;
  const h16* gf = g + (long)(C + c) * DHW + hw;
  h16* o = out + (long)c * DHW + hw;
  float h = 0.f;
  for (int t = 0; t < D; ++t) {
    int d = reverse ? (D - 1 - t) : t;
    float z = tanhf((float)gz[(long)d * HW]);
    float f = 1.f / (1.f + expf(-(float)gf[(long)d * HW]));
    h = f * h + (1.f - f) * z;
    o[(long)d * HW] = (h16)h;
  }
}

// Bidirectional 3-gate scan: gates [3C][D][HW]: z, f_fwd, f_bwd. Out = fwd+bwd.
__global__ void k_qrnn_bi3(const h16* __restrict__ g, h16* __restrict__ out,
                           int C, int D, long HW) {
  long i = (long)blockIdx.x * blockDim.x + threadIdx.x;
  if (i >= (long)C * HW) return;
  int c = (int)(i / HW);
  long hw = i % HW;
  long DHW = (long)D * HW;
  const h16* gz = g + (long)c * DHW + hw;
  const h16* g1 = g + (long)(C + c) * DHW + hw;
  const h16* g2 = g + (long)(2 * C + c) * DHW + hw;
  h16* o = out + (long)c * DHW + hw;
  float h = 0.f;
  for (int d = 0; d < D; ++d) {
    float z = tanhf((float)gz[(long)d * HW]);
    float f = 1.f / (1.f + expf(-(float)g1[(long)d * HW]));
    h = f * h + (1.f - f) * z;
    o[(long)d * HW] = (h16)h;
  }
  h = 0.f;
  for (int d = D - 1; d >= 0; --d) {
    float z = tanhf((float)gz[(long)d * HW]);
    float f = 1.f / (1.f + expf(-(float)g2[(long)d * HW]));
    h = f * h + (1.f - f) * z;
    o[(long)d * HW] = (h16)((float)o[(long)d * HW] + h);
  }
}

// ---------------------------------------------------------------------------
// x2 bilinear upsample on H,W with align_corners=True. out [C][D][2H][2W].
// ---------------------------------------------------------------------------
__global__ void k_up2(const h16* __restrict__ in, h16* __restrict__ out,
                      int C, int D, int H, int W) {
  int Ho = 2 * H, Wo = 2 * W;
  long N = (long)C * D * Ho * Wo;
  long i = (long)blockIdx.x * blockDim.x + threadIdx.x;
  if (i >= N) return;
  int ow = (int)(i % Wo);
  long r = i / Wo;
  int oh = (int)(r % Ho);
  r /= Ho;
  int d = (int)(r % D);
  int c = (int)(r / D);
  float ph = oh * ((float)(H - 1) / (float)(Ho - 1));
  float pw = ow * ((float)(W - 1) / (float)(Wo - 1));
  int hl = (int)ph; int hh = min(hl + 1, H - 1); float th = ph - hl;
  int wl = (int)pw; int wh = min(wl + 1, W - 1); float tw = pw - wl;
  const h16* b = in + ((long)c * D + d) * H * W;
  float v00 = (float)b[(long)hl * W + wl], v01 = (float)b[(long)hl * W + wh];
  float v10 = (float)b[(long)hh * W + wl], v11 = (float)b[(long)hh * W + wh];
  float v = (1.f - th) * ((1.f - tw) * v00 + tw * v01) +
            th * ((1.f - tw) * v10 + tw * v11);
  out[i] = (h16)v;
}

// ---------------------------------------------------------------------------
// Host orchestration
// ---------------------------------------------------------------------------
namespace {

struct Bump {
  char* p; size_t off;
  void* get(size_t bytes) {
    void* r = p + off;
    off += (bytes + 255) & ~(size_t)255;
    return r;
  }
};

struct Conv { h16* A; int Cout, Cin, K3, Mpad, Kpad, Kreal; };

inline unsigned gb(long n) { return (unsigned)((n + 255) / 256); }

Conv prep(Bump& ws, const float* w, int O, int I, int K3, int deconv, hipStream_t s) {
  Conv c;
  c.Cout = O; c.Cin = I; c.K3 = K3;
  c.Kreal = I * K3 * K3 * K3;
  c.Kpad  = ((c.Kreal + 31) / 32) * 32;
  c.Mpad  = ((O + 63) / 64) * 64;
  c.A = (h16*)ws.get((size_t)c.Mpad * c.Kpad * sizeof(h16));
  long tot = (long)c.Mpad * c.Kpad;
  k_pack_w<<<gb(tot), 256, 0, s>>>(w, c.A, O, I, K3, c.Mpad, c.Kpad, deconv);
  return c;
}

void run_conv(const Conv& c, const h16* in, int Din, int Hin, int Win,
              h16* out, int Do, int Ho, int Wo, int sHW,
              const float* bias, hipStream_t s) {
  long Ntot = (long)Do * Ho * Wo;
  dim3 grid((unsigned)((Ntot + TN - 1) / TN), (unsigned)(c.Mpad / TM));
  if (c.K3 == 3)
    k_conv_wmma<3, 1><<<grid, 128, 0, s>>>(in, c.A, bias, out, Din, Hin, Win,
                                           c.Cout, Do, Ho, Wo, sHW, sHW,
                                           c.Kreal, c.Kpad);
  else
    k_conv_wmma<1, 0><<<grid, 128, 0, s>>>(in, c.A, bias, out, Din, Hin, Win,
                                           c.Cout, Do, Ho, Wo, sHW, sHW,
                                           c.Kreal, c.Kpad);
}

void qrnn2(const h16* g, h16* o, int C, int D, long HW, int rev, hipStream_t s) {
  long n = (long)C * HW;
  k_qrnn2<<<gb(n), 256, 0, s>>>(g, o, C, D, HW, rev);
}

} // namespace

extern "C" void kernel_launch(void* const* d_in, const int* in_sizes, int n_in,
                              void* d_out, int out_size, void* d_ws, size_t ws_size,
                              hipStream_t stream) {
  (void)in_sizes; (void)n_in; (void)out_size; (void)ws_size;
  const float* x = (const float*)d_in[0];
  // sorted(PARAM_SHAPES) index map
  const float* W_c2_1 = (const float*)d_in[1];
  const float* W_c2_2 = (const float*)d_in[2];
  const float* W_c2_r = (const float*)d_in[3];
  const float* W_c3_1 = (const float*)d_in[4];
  const float* W_c3_2 = (const float*)d_in[5];
  const float* W_c3_r = (const float*)d_in[6];
  const float* W_c4_1 = (const float*)d_in[7];
  const float* W_c4_2 = (const float*)d_in[8];
  const float* W_c4_r = (const float*)d_in[9];
  const float* W_c5_1 = (const float*)d_in[10];
  const float* W_c5_2 = (const float*)d_in[11];
  const float* W_c5_r = (const float*)d_in[12];
  const float* W_conv1 = (const float*)d_in[13];
  const float* W_down1 = (const float*)d_in[14];
  const float* W_down2 = (const float*)d_in[15];
  const float* W_down3 = (const float*)d_in[16];
  const float* W_down4 = (const float*)d_in[17];
  const float* B_final = (const float*)d_in[18];
  const float* W_final = (const float*)d_in[19];
  const float* W_u2_1 = (const float*)d_in[20];
  const float* W_u2_2 = (const float*)d_in[21];
  const float* W_u2_r = (const float*)d_in[22];
  const float* W_u3_1 = (const float*)d_in[23];
  const float* W_u3_2 = (const float*)d_in[24];
  const float* W_u3_r = (const float*)d_in[25];
  const float* W_u4_1 = (const float*)d_in[26];
  const float* W_u4_2 = (const float*)d_in[27];
  const float* W_u4_r = (const float*)d_in[28];
  const float* W_u5_1 = (const float*)d_in[29];
  const float* W_u5_2 = (const float*)d_in[30];
  const float* W_u5_r = (const float*)d_in[31];
  const float* W_up2 = (const float*)d_in[32];
  const float* W_up3 = (const float*)d_in[33];
  const float* W_up4 = (const float*)d_in[34];
  const float* W_up5 = (const float*)d_in[35];
  float* out = (float*)d_out;

  const int D = 31;
  const int HH[5] = {192, 96, 48, 24, 12};
  long HWl[5], VOX[5];
  for (int l = 0; l < 5; ++l) { HWl[l] = (long)HH[l] * HH[l]; VOX[l] = (long)D * HWl[l]; }
  const long V1 = VOX[0];

  Bump ws{ (char*)d_ws, 0 };

  // --- weights (repacked each call; deterministic) ---
  Conv c1   = prep(ws, W_conv1, 48, 1, 3, 0, stream);
  Conv cdn[4] = { prep(ws, W_down1, 32, 16, 3, 0, stream),
                  prep(ws, W_down2, 64, 32, 3, 0, stream),
                  prep(ws, W_down3, 128, 64, 3, 0, stream),
                  prep(ws, W_down4, 256, 128, 3, 0, stream) };
  Conv cb1[4] = { prep(ws, W_c2_1, 64, 16, 3, 0, stream),
                  prep(ws, W_c3_1, 128, 32, 3, 0, stream),
                  prep(ws, W_c4_1, 256, 64, 3, 0, stream),
                  prep(ws, W_c5_1, 512, 128, 3, 0, stream) };
  Conv cb2[4] = { prep(ws, W_c2_2, 64, 32, 3, 0, stream),
                  prep(ws, W_c3_2, 128, 64, 3, 0, stream),
                  prep(ws, W_c4_2, 256, 128, 3, 0, stream),
                  prep(ws, W_c5_2, 512, 256, 3, 0, stream) };
  Conv cbr[4] = { prep(ws, W_c2_r, 64, 16, 1, 0, stream),
                  prep(ws, W_c3_r, 128, 32, 1, 0, stream),
                  prep(ws, W_c4_r, 256, 64, 1, 0, stream),
                  prep(ws, W_c5_r, 512, 128, 1, 0, stream) };
  Conv cup[4] = { prep(ws, W_up5, 256, 256, 3, 0, stream),
                  prep(ws, W_up4, 128, 128, 3, 0, stream),
                  prep(ws, W_up3, 64, 64, 3, 0, stream),
                  prep(ws, W_up2, 32, 32, 3, 0, stream) };
  Conv cu1[4] = { prep(ws, W_u5_1, 256, 256, 3, 1, stream),
                  prep(ws, W_u4_1, 128, 128, 3, 1, stream),
                  prep(ws, W_u3_1, 64, 64, 3, 1, stream),
                  prep(ws, W_u2_1, 32, 32, 3, 1, stream) };
  Conv cu2[4] = { prep(ws, W_u5_2, 256, 128, 3, 1, stream),
                  prep(ws, W_u4_2, 128, 64, 3, 1, stream),
                  prep(ws, W_u3_2, 64, 32, 3, 1, stream),
                  prep(ws, W_u2_2, 32, 16, 3, 1, stream) };
  Conv cuR[4] = { prep(ws, W_u5_r, 256, 256, 1, 1, stream),
                  prep(ws, W_u4_r, 128, 128, 1, 1, stream),
                  prep(ws, W_u3_r, 64, 64, 1, 1, stream),
                  prep(ws, W_u2_r, 32, 32, 1, 1, stream) };
  Conv cfin = prep(ws, W_final, 3, 16, 3, 1, stream);

  // --- activation buffers (f16) ---
  h16* x16 = (h16*)ws.get((size_t)V1 * 2);
  h16* e[5];
  int ech[5] = {16, 32, 64, 128, 256};
  for (int l = 0; l < 5; ++l) e[l] = (h16*)ws.get((size_t)VOX[l] * ech[l] * 2);
  h16* GT0 = (h16*)ws.get((size_t)V1 * 48 * 2);   // gate scratch, max 48ch @ L1
  h16* SA  = (h16*)ws.get((size_t)V1 * 32 * 2);   // upsampled-input scratch
  h16* SB  = (h16*)ws.get((size_t)V1 * 32 * 2);   // concat scratch
  h16* HA  = (h16*)ws.get((size_t)V1 * 16 * 2);
  h16* HB  = (h16*)ws.get((size_t)V1 * 16 * 2);
  h16* HC  = (h16*)ws.get((size_t)V1 * 16 * 2);
  h16* o16 = (h16*)ws.get((size_t)V1 * 2);

  // ===== encoder =====
  k_f32_to_f16<<<gb(V1), 256, 0, stream>>>(x, x16, V1);

  // e1 = bi-QRNN(conv1(x)) : 48 gates -> 16 ch
  run_conv(c1, x16, D, HH[0], HH[0], GT0, D, HH[0], HH[0], 1, nullptr, stream);
  k_qrnn_bi3<<<gb(16 * HWl[0]), 256, 0, stream>>>(GT0, e[0], 16, D, HWl[0]);

  for (int i = 0; i < 4; ++i) {
    int li = i, lo = i + 1;
    int pc = 16 << i;                     // input channels at level li
    // down-conv stride(1,2,2), reverse scan
    run_conv(cdn[i], e[i], D, HH[li], HH[li], GT0, D, HH[lo], HH[lo], 2, nullptr, stream);
    qrnn2(GT0, HA, pc, D, HWl[lo], 1, stream);
    // conv_block (reverse=False)
    run_conv(cb1[i], HA, D, HH[lo], HH[lo], GT0, D, HH[lo], HH[lo], 1, nullptr, stream);
    qrnn2(GT0, HB, 2 * pc, D, HWl[lo], 0, stream);
    run_conv(cb2[i], HB, D, HH[lo], HH[lo], GT0, D, HH[lo], HH[lo], 1, nullptr, stream);
    qrnn2(GT0, HC, 2 * pc, D, HWl[lo], 0, stream);
    run_conv(cbr[i], HA, D, HH[lo], HH[lo], GT0, D, HH[lo], HH[lo], 1, nullptr, stream);
    qrnn2(GT0, HB, 2 * pc, D, HWl[lo], 0, stream);
    long n = (long)2 * pc * VOX[lo];
    k_add_f16<<<gb(n), 256, 0, stream>>>(HC, HB, e[i + 1], n);
  }

  // ===== decoder =====
  int sch[4] = {128, 64, 32, 16};
  h16* skips[4] = { e[3], e[2], e[1], e[0] };
  for (int i = 0; i < 4; ++i) {
    int ls = 4 - i, ld = 3 - i;
    int sc = sch[i];                      // skip channels; input has 2*sc
    const h16* din = (i == 0) ? e[4] : HA;
    // x2 upsample on H,W then up-conv, reverse scan
    k_up2<<<gb((long)2 * sc * VOX[ls] * 4), 256, 0, stream>>>(din, SA, 2 * sc, D, HH[ls], HH[ls]);
    run_conv(cup[i], SA, D, HH[ld], HH[ld], GT0, D, HH[ld], HH[ld], 1, nullptr, stream);
    qrnn2(GT0, HA, sc, D, HWl[ld], 1, stream);
    // concat [skip, d]
    size_t half = (size_t)sc * VOX[ld] * 2;
    hipMemcpyAsync(SB, skips[i], half, hipMemcpyDeviceToDevice, stream);
    hipMemcpyAsync((char*)SB + half, HA, half, hipMemcpyDeviceToDevice, stream);
    // deconv_block (reverse=False)
    run_conv(cu1[i], SB, D, HH[ld], HH[ld], GT0, D, HH[ld], HH[ld], 1, nullptr, stream);
    qrnn2(GT0, HB, sc, D, HWl[ld], 0, stream);
    run_conv(cu2[i], HB, D, HH[ld], HH[ld], GT0, D, HH[ld], HH[ld], 1, nullptr, stream);
    qrnn2(GT0, HC, sc, D, HWl[ld], 0, stream);
    run_conv(cuR[i], SB, D, HH[ld], HH[ld], GT0, D, HH[ld], HH[ld], 1, nullptr, stream);
    qrnn2(GT0, HB, sc, D, HWl[ld], 0, stream);
    long n = (long)sc * VOX[ld];
    k_add_f16<<<gb(n), 256, 0, stream>>>(HC, HB, HA, n);
  }

  // ===== final: deconv(16->3,+bias) -> bi-QRNN(1ch) -> + x =====
  run_conv(cfin, HA, D, HH[0], HH[0], GT0, D, HH[0], HH[0], 1, B_final, stream);
  k_qrnn_bi3<<<gb(HWl[0]), 256, 0, stream>>>(GT0, o16, 1, D, HWl[0]);
  k_final_add<<<gb(V1), 256, 0, stream>>>(o16, x, out, V1);
}
